// GumbelVectorQuantizer_56513179681191
// MI455X (gfx1250) — matrix operand order repaired and checked by reading, not compiled, not run on hardware
//
#include <hip/hip_runtime.h>

typedef _Float16 half16v __attribute__((ext_vector_type(16)));
typedef _Float16 half8v  __attribute__((ext_vector_type(8)));
typedef _Float16 half4v  __attribute__((ext_vector_type(4)));
typedef float    f8v     __attribute__((ext_vector_type(8)));

#define NTOK   16384          // B*S
#define DIM    1024           // D
#define NGRP   2              // G
#define VSZ    320            // V
#define DGRP   512            // D/G
#define NCOL   640            // G*V
#define MTILE  128
#define LDSA_STRIDE 1032      // 1024 + 8 halves pad (bank spread for TR-style loads)
#define OUT_IDX_OFF  ((size_t)NTOK * DIM)              // 16777216 floats of "out"
#define OUT_LOSS_OFF (OUT_IDX_OFF + (size_t)NTOK * NGRP)

// --- kernel 0: W f32 -> f16 into workspace, zero histogram ------------------
__global__ void vq_prep_kernel(const float* __restrict__ W,
                               _Float16* __restrict__ W16,
                               float* __restrict__ hist) {
  int t = blockIdx.x * blockDim.x + threadIdx.x;     // one float4 per thread
  if (t < (NCOL * DIM) / 4) {
    float4 v = ((const float4*)W)[t];
    half4v h;
    h[0] = (_Float16)v.x; h[1] = (_Float16)v.y;
    h[2] = (_Float16)v.z; h[3] = (_Float16)v.w;
    ((half4v*)W16)[t] = h;
  }
  if (t < NCOL) hist[t] = 0.0f;
}

// --- kernel 1: fused GEMM(WMMA f16) + argmax + histogram + gather -----------
__global__ void __launch_bounds__(256, 1)
vq_gemm_argmax_gather(const float* __restrict__ hs,
                      const _Float16* __restrict__ W16,
                      const float* __restrict__ bvec,
                      const float* __restrict__ cv,
                      float* __restrict__ out,
                      float* __restrict__ hist) {
  extern __shared__ char smem[];
  _Float16* ldsA   = (_Float16*)smem;                         // [MTILE][LDSA_STRIDE] f16
  int*      ldsIdx = (int*)(smem + MTILE * LDSA_STRIDE * 2);  // [MTILE][NGRP]

  const int rowBase = blockIdx.x * MTILE;
  const int tid     = threadIdx.x;
  const int wave    = tid >> 5;
  const int lane    = tid & 31;
  const int lane16  = lane & 15;
  const int hi      = lane >> 4;

  // Stage 128x1024 hidden-state tile into LDS as f16 (converted in flight).
  for (int i = tid; i < MTILE * DIM / 4; i += 256) {
    int row = i >> 8;              // 256 float4 per row
    int c4  = i & 255;
    float4 v = ((const float4*)(hs + (size_t)(rowBase + row) * DIM))[c4];
    half4v h;
    h[0] = (_Float16)v.x; h[1] = (_Float16)v.y;
    h[2] = (_Float16)v.z; h[3] = (_Float16)v.w;
    *(half4v*)(ldsA + row * LDSA_STRIDE + c4 * 4) = h;
  }
  __syncthreads();

  const int mrow = wave * 16;                                  // wave's 16-row subtile
  const _Float16* aRow = ldsA + (size_t)(mrow + lane16) * LDSA_STRIDE;
  const int cbase = hi ? 16 : 0;                               // contiguous K: half h holds k = 16h+s, as B

  for (int g = 0; g < NGRP; ++g) {
    float bestv[8]; int besti[8];
#pragma unroll
    for (int v = 0; v < 8; ++v) { bestv[v] = -__builtin_inff(); besti[v] = 0; }

    for (int nt = 0; nt < VSZ / 16; ++nt) {
      const int ncol = g * VSZ + nt * 16 + lane16;             // lane's W row (B column)
      const _Float16* bRow = W16 + (size_t)ncol * DIM + hi * 16;
      f8v acc = {0.f, 0.f, 0.f, 0.f, 0.f, 0.f, 0.f, 0.f};
#pragma unroll 4
      for (int k0 = 0; k0 < DIM; k0 += 32) {
        half8v alo = *(const half8v*)(aRow + k0 + cbase);
        half8v ahi = *(const half8v*)(aRow + k0 + cbase + 8);
        half16v a = __builtin_shufflevector(alo, ahi,
                      0,1,2,3,4,5,6,7,8,9,10,11,12,13,14,15);
        half16v bm = *(const half16v*)(bRow + k0);
        acc = __builtin_amdgcn_wmma_f32_16x16x32_f16(
                  false, a, false, bm, (short)0, acc, false, false);
      }
      const float bias = bvec[g * VSZ + nt * 16 + lane16];
      const int nloc = nt * 16 + lane16;
#pragma unroll
      for (int v = 0; v < 8; ++v) {
        float val = acc[v] + bias;
        if (val > bestv[v]) { bestv[v] = val; besti[v] = nloc; }  // strict >: first max
      }
    }

    // Cross-lane argmax within each 16-lane half (xor<16 never crosses halves).
    int myIdx = 0;
#pragma unroll
    for (int v = 0; v < 8; ++v) {
      float val = bestv[v]; int idx = besti[v];
#pragma unroll
      for (int m = 1; m < 16; m <<= 1) {
        float ov = __shfl_xor(val, m, 32);
        int   oi = __shfl_xor(idx, m, 32);
        if (ov > val || (ov == val && oi < idx)) { val = ov; idx = oi; }
      }
      if (lane16 == v) myIdx = idx;
    }
    if (lane16 < 8) {
      int rl = mrow + lane16 + 8 * hi;                 // local row (C layout: M = v + 8*hi)
      ldsIdx[rl * NGRP + g] = myIdx;
      size_t rglob = (size_t)(rowBase + rl);
      out[OUT_IDX_OFF + rglob * NGRP + g] = (float)myIdx;
      atomicAdd(&hist[g * VSZ + myIdx], 1.0f);
    }
  }
  __syncthreads();

  // Gather codevectors: 256 (row,group) pairs, 8 waves, float4 copies.
  for (int p = wave; p < MTILE * NGRP; p += 8) {
    int rl = p >> 1, g = p & 1;
    int idx = ldsIdx[rl * NGRP + g];
    const float4* src = (const float4*)(cv + (size_t)(g * VSZ + idx) * DGRP);
    float4* dst = (float4*)(out + (size_t)(rowBase + rl) * DIM + g * DGRP);
#pragma unroll
    for (int i = 0; i < DGRP / 4 / 32; ++i)
      dst[lane + i * 32] = src[lane + i * 32];
  }
}

// --- kernel 2: perplexity -> diversity loss ---------------------------------
__global__ void vq_finalize_kernel(const float* __restrict__ hist,
                                   float* __restrict__ out) {
  __shared__ float part[64];
  int t = threadIdx.x;
  float perp = 0.0f;
  for (int g = 0; g < NGRP; ++g) {
    float s = 0.0f;
    for (int v = t; v < VSZ; v += 64) {
      float m = hist[g * VSZ + v] * (1.0f / (float)NTOK);
      s += m * __logf(m + 1e-7f);
    }
    part[t] = s;
    __syncthreads();
    if (t == 0) {
      float tot = 0.0f;
      for (int i = 0; i < 64; ++i) tot += part[i];
      perp += __expf(-tot);
    }
    __syncthreads();
  }
  if (t == 0) out[OUT_LOSS_OFF] = ((float)NCOL - perp) / (float)NCOL * 0.1f;
}

extern "C" void kernel_launch(void* const* d_in, const int* in_sizes, int n_in,
                              void* d_out, int out_size, void* d_ws, size_t ws_size,
                              hipStream_t stream) {
  const float* hs   = (const float*)d_in[0];   // [16384,1024]
  const float* W    = (const float*)d_in[1];   // [640,1024]
  const float* bvec = (const float*)d_in[2];   // [640]
  const float* cv   = (const float*)d_in[3];   // [640,512]
  float* out = (float*)d_out;

  _Float16* W16 = (_Float16*)d_ws;                                    // 1,310,720 B
  float* hist = (float*)((char*)d_ws + (size_t)NCOL * DIM * sizeof(_Float16));

  vq_prep_kernel<<<(NCOL * DIM / 4 + 255) / 256, 256, 0, stream>>>(W, W16, hist);

  size_t shmem = (size_t)MTILE * LDSA_STRIDE * sizeof(_Float16)
               + (size_t)MTILE * NGRP * sizeof(int);                  // ~265 KB (<320 KB/WGP)
  vq_gemm_argmax_gather<<<NTOK / MTILE, 256, shmem, stream>>>(
      hs, W16, bvec, cv, out, hist);

  vq_finalize_kernel<<<1, 64, 0, stream>>>(hist, out);
}
